// PatchedSwinSelfAttention_54047868453199
// MI455X (gfx1250) — matrix operand order repaired and checked by reading, not compiled, not run on hardware
//
#include <hip/hip_runtime.h>

typedef __attribute__((ext_vector_type(16))) _Float16 v16h;
typedef __attribute__((ext_vector_type(8)))  float    v8f;

#define NW 49     // window size (tokens)
#define NP 64     // padded tokens
#define CC 128    // channels
#define NH 4      // heads
#define DH 32     // head dim
#define LQ 136    // padded LDS row stride (halves) for Qs/Ks  (bank-conflict free)
#define LV 72     // padded LDS row stride (halves) for Vts/Ps

union AFrag { v16h v; _Float16 h[16]; unsigned u[8]; };

__device__ inline unsigned pack2h(float a, float b) {
  _Float16 ha = (_Float16)a, hb = (_Float16)b;
  unsigned short ua = __builtin_bit_cast(unsigned short, ha);
  unsigned short ub = __builtin_bit_cast(unsigned short, hb);
  return (unsigned)ua | ((unsigned)ub << 16);
}

// Pack W^T (B[k][n] = W[n][k]) into CDNA5 f16 B-fragment order:
// packed[((p*4+kt)*8+nt)*32 + lane][j] = {W[n][k], W[n][k+1]},
// n = nt*16 + lane%16, k = kt*32 + (j/4)*16 + (lane/16)*8 + 2*(j%4).
__global__ void pack_w(const float* __restrict__ Wq, const float* __restrict__ Wk,
                       const float* __restrict__ Wv, unsigned* __restrict__ packed) {
  int t = blockIdx.x * blockDim.x + threadIdx.x;
  if (t >= 3 * 32 * 32 * 8) return;
  int j    =  t        & 7;
  int lane = (t >> 3)  & 31;
  int tile = (t >> 8)  & 31;
  int p    =  t >> 13;
  const float* W = (p == 0) ? Wq : ((p == 1) ? Wk : Wv);
  int kt = tile >> 3, nt = tile & 7;
  int n  = nt * 16 + (lane & 15);
  int k  = kt * 32 + (j >> 2) * 16 + (lane >> 4) * 8 + 2 * (j & 3);
  packed[t] = pack2h(W[n * CC + k], W[n * CC + k + 1]);
}

// biasFull[h][m][n] = table[idx[m*49+n]][h]
__global__ void build_bias(const float* __restrict__ table, const int* __restrict__ idx,
                           float* __restrict__ biasFull) {
  int t = blockIdx.x * blockDim.x + threadIdx.x;
  if (t >= NW * NW) return;
  int id = idx[t];
  for (int h = 0; h < NH; ++h)
    biasFull[h * NW * NW + t] = table[id * NH + h];
}

__global__ __launch_bounds__(128) void swin_attn(
    const float* __restrict__ hs, const float* __restrict__ mask,
    const unsigned* __restrict__ packed,
    const float* __restrict__ bq, const float* __restrict__ bk, const float* __restrict__ bv,
    const float* __restrict__ biasFull, float* __restrict__ out) {
  __shared__ __align__(16) _Float16 Qs[NP * LQ];      // [token][c]  (pre-scaled by 1/sqrt(DH))
  __shared__ __align__(16) _Float16 Ks[NP * LQ];      // [token][c]
  __shared__ __align__(16) _Float16 Vts[CC * LV];     // [c][token]  (transposed)
  __shared__ __align__(16) _Float16 Ps[NH][16 * LV];  // per-wave prob slice [row][token]

  const int b    = blockIdx.x;
  const int lane = threadIdx.x & 31;
  const int wave = threadIdx.x >> 5;
  const int mrow = lane & 15;
  const int half = lane >> 4;

  // ---- Phase 1: X A-fragments for this wave's 16 token rows (zeros past row 49) ----
  AFrag ax[4];
  {
    const int m = wave * 16 + mrow;
    const float* row = hs + ((size_t)b * NW + m) * CC;
#pragma unroll
    for (int kt = 0; kt < 4; ++kt)
#pragma unroll
      for (int j = 0; j < 8; ++j) {
        int k = kt * 32 + (j >> 2) * 16 + half * 8 + (j & 3) * 2;
        float x0 = 0.f, x1 = 0.f;
        if (m < NW) { x0 = row[k]; x1 = row[k + 1]; }
        ax[kt].u[j] = pack2h(x0, x1);
      }
  }

  // ---- Phase 2: Q/K/V projections (bias injected through C operand) ----
  const float qscale = 0.17677669529663687f;  // 1/sqrt(32)
  for (int p = 0; p < 3; ++p) {
    const float* bias = (p == 0) ? bq : ((p == 1) ? bk : bv);
#pragma unroll
    for (int nt = 0; nt < 8; ++nt) {
      float bval = bias[nt * 16 + mrow];
      v8f acc;
#pragma unroll
      for (int r = 0; r < 8; ++r) acc[r] = bval;
#pragma unroll
      for (int kt = 0; kt < 4; ++kt) {
        const v16h bf = *(const v16h*)(packed + ((size_t)((p * 4 + kt) * 8 + nt) * 32 + lane) * 8);
        acc = __builtin_amdgcn_wmma_f32_16x16x32_f16(false, ax[kt].v, false, bf,
                                                     (short)0, acc, false, false);
      }
#pragma unroll
      for (int r = 0; r < 8; ++r) {
        int m = wave * 16 + half * 8 + r;
        int col = nt * 16 + mrow;
        float v = acc[r];
        if (p == 0)      Qs[m * LQ + col]  = (_Float16)(v * qscale);
        else if (p == 1) Ks[m * LQ + col]  = (_Float16)v;
        else             Vts[col * LV + m] = (_Float16)v;
      }
    }
  }
  __syncthreads();

  // ---- Phase 3: attention, head h = wave ----
  const int h = wave;
  const float* maskB = mask + (size_t)b * NW * NW;
  const float* biasH = biasFull + h * NW * NW;

  for (int mt = 0; mt < 4; ++mt) {
    // Q A-fragment (16 token rows, K = 32 head dims)
    AFrag aq;
    {
      const _Float16* qrow = &Qs[(mt * 16 + mrow) * LQ + h * DH];
      *(uint4*)&aq.u[0] = *(const uint4*)(qrow + half * 8);
      *(uint4*)&aq.u[4] = *(const uint4*)(qrow + 16 + half * 8);
    }
    v8f s[4];
#pragma unroll
    for (int nt = 0; nt < 4; ++nt) {
#pragma unroll
      for (int r = 0; r < 8; ++r) {
        int m = mt * 16 + half * 8 + r;
        int n = nt * 16 + mrow;
        float init;
        if (n >= NW)     init = -1e30f;
        else if (m < NW) init = biasH[m * NW + n] + maskB[m * NW + n];
        else             init = 0.f;
        s[nt][r] = init;
      }
      AFrag bkf;  // K^T B-fragment: K = head dims (VGPRs), N = tokens (lanes)
      {
        const _Float16* krow = &Ks[(nt * 16 + mrow) * LQ + h * DH + half * 8];
        *(uint4*)&bkf.u[0] = *(const uint4*)(krow);
        *(uint4*)&bkf.u[4] = *(const uint4*)(krow + 16);
      }
      s[nt] = __builtin_amdgcn_wmma_f32_16x16x32_f16(false, aq.v, false, bkf.v,
                                                     (short)0, s[nt], false, false);
    }
    // row softmax (row = 16-lane group; columns = 4 fragments x 16 lanes)
#pragma unroll
    for (int r = 0; r < 8; ++r) {
      float mx = fmaxf(fmaxf(s[0][r], s[1][r]), fmaxf(s[2][r], s[3][r]));
#pragma unroll
      for (int off = 8; off; off >>= 1) mx = fmaxf(mx, __shfl_xor(mx, off, 32));
      float e0 = __expf(s[0][r] - mx), e1 = __expf(s[1][r] - mx);
      float e2 = __expf(s[2][r] - mx), e3 = __expf(s[3][r] - mx);
      float sm = e0 + e1 + e2 + e3;
#pragma unroll
      for (int off = 8; off; off >>= 1) sm += __shfl_xor(sm, off, 32);
      float inv = 1.0f / sm;
      s[0][r] = e0 * inv; s[1][r] = e1 * inv; s[2][r] = e2 * inv; s[3][r] = e3 * inv;
    }
    // probs -> f16 LDS slice (per-wave)
    _Float16* Pw = Ps[wave];
#pragma unroll
    for (int nt = 0; nt < 4; ++nt)
#pragma unroll
      for (int r = 0; r < 8; ++r)
        Pw[(half * 8 + r) * LV + nt * 16 + mrow] = (_Float16)s[nt][r];
    __syncthreads();

    // P x V
    AFrag ap[2];
#pragma unroll
    for (int kt2 = 0; kt2 < 2; ++kt2) {
      const _Float16* prow = &Pw[mrow * LV + kt2 * 32];
      *(uint4*)&ap[kt2].u[0] = *(const uint4*)(prow + half * 8);
      *(uint4*)&ap[kt2].u[4] = *(const uint4*)(prow + 16 + half * 8);
    }
#pragma unroll
    for (int dt = 0; dt < 2; ++dt) {
      v8f c = {};
#pragma unroll
      for (int kt2 = 0; kt2 < 2; ++kt2) {
        AFrag bv2;  // V B-fragment from transposed LDS: K = tokens contiguous
        const _Float16* vrow = &Vts[(h * DH + dt * 16 + mrow) * LV + kt2 * 32 + half * 8];
        *(uint4*)&bv2.u[0] = *(const uint4*)(vrow);
        *(uint4*)&bv2.u[4] = *(const uint4*)(vrow + 16);
        c = __builtin_amdgcn_wmma_f32_16x16x32_f16(false, ap[kt2].v, false, bv2.v,
                                                   (short)0, c, false, false);
      }
#pragma unroll
      for (int r = 0; r < 8; ++r) {
        int m = mt * 16 + half * 8 + r;
        if (m < NW)
          out[((size_t)b * NW + m) * CC + h * DH + dt * 16 + mrow] = c[r];
      }
    }
  }
}

extern "C" void kernel_launch(void* const* d_in, const int* in_sizes, int n_in,
                              void* d_out, int out_size, void* d_ws, size_t ws_size,
                              hipStream_t stream) {
  (void)in_sizes; (void)n_in; (void)out_size; (void)ws_size;
  const float* hs    = (const float*)d_in[0];
  const float* mask  = (const float*)d_in[1];
  const float* Wq    = (const float*)d_in[2];
  const float* bq    = (const float*)d_in[3];
  const float* Wk    = (const float*)d_in[4];
  const float* bk    = (const float*)d_in[5];
  const float* Wv    = (const float*)d_in[6];
  const float* bv    = (const float*)d_in[7];
  const float* table = (const float*)d_in[8];
  const int*   idx   = (const int*)d_in[9];

  unsigned* packed  = (unsigned*)d_ws;                          // 3*32*32*8 u32 = 96 KB
  float*    biasF   = (float*)((char*)d_ws + 3 * 32 * 32 * 8 * 4);  // 4*49*49 f32

  pack_w<<<(3 * 32 * 32 * 8 + 255) / 256, 256, 0, stream>>>(Wq, Wk, Wv, packed);
  build_bias<<<(NW * NW + 255) / 256, 256, 0, stream>>>(table, idx, biasF);
  swin_attn<<<4096, 128, 0, stream>>>(hs, mask, packed, bq, bk, bv, biasF, (float*)d_out);
}
